// SoftPoolFeat_15470472200290
// MI455X (gfx1250) — matrix operand order repaired and checked by reading, not compiled, not run on hardware
//
#include <hip/hip_runtime.h>
#include <hip/hip_bf16.h>

// ---------------------------------------------------------------------------
// SoftPoolFeat for MI455X (gfx1250).
// Memory-bound: 272 MB of feature output dominates (~12us floor @ 23.3 TB/s).
// GEMMs (~2.5 GFLOP) use v_wmma_f32_16x16x32_f16 (wave32 WMMA), K templated
// so fragment loads are unconditional (no per-element EXEC branching).
// The gather/emit stage uses async global->LDS staging when available.
// ---------------------------------------------------------------------------

#define Bv   2
#define CIN  15
#define NPTS 2048
#define RREG 16
#define FD   256
#define NCAB 16
#define PCAB (NPTS / NCAB)
#define EPSV 1e-5f

typedef __attribute__((ext_vector_type(16))) _Float16 v16h;
typedef __attribute__((ext_vector_type(8)))  float    v8f;

#ifndef __has_builtin
#define __has_builtin(x) 0
#endif

#if __has_builtin(__builtin_amdgcn_global_load_async_to_lds_b128) && \
    __has_builtin(__builtin_amdgcn_s_wait_asynccnt)
#define HAVE_ASYNC_LDS 1
#endif

#ifdef HAVE_ASYNC_LDS
typedef __attribute__((__vector_size__(4 * sizeof(int)))) int i32x4;
typedef __attribute__((address_space(1))) i32x4 gvec_t;
typedef __attribute__((address_space(3))) i32x4 lvec_t;
#endif

// Cooperatively stage a 2048-float (8KB) global row into LDS.
// With async path: 2 x b128 per thread (256 threads x 2 x 16B = 8KB).
__device__ __forceinline__ void stage_row2048(const float* __restrict__ g, float* lds) {
#ifdef HAVE_ASYNC_LDS
  // AS casts via integers (global VA == flat VA; LDS offset == flat[31:0]).
  const unsigned long long gv = (unsigned long long)(size_t)g;
  const unsigned int       lv = (unsigned int)(size_t)lds;
  const int t = (int)threadIdx.x;
  gvec_t* gp0 = (gvec_t*)(gv + (unsigned long long)t * 16);
  lvec_t* lp0 = (lvec_t*)(size_t)(lv + (unsigned)t * 16);
  __builtin_amdgcn_global_load_async_to_lds_b128(gp0, lp0, 0, 0);
  gvec_t* gp1 = (gvec_t*)(gv + 4096ull + (unsigned long long)t * 16);
  lvec_t* lp1 = (lvec_t*)(size_t)(lv + 4096u + (unsigned)t * 16);
  __builtin_amdgcn_global_load_async_to_lds_b128(gp1, lp1, 0, 0);
#else
  for (int i = (int)threadIdx.x; i < 2048; i += (int)blockDim.x) lds[i] = g[i];
#endif
}

__device__ __forceinline__ void wait_stage_and_barrier() {
#ifdef HAVE_ASYNC_LDS
  __builtin_amdgcn_s_wait_asynccnt(0);
#endif
  __syncthreads();
}

// ---------------------------------------------------------------------------
// WMMA GEMM: Y[b,m,n] = sum_k W[m,k] * X[b,k,n] + bias[m]
// One wave per 16x16 output tile. K is a compile-time constant:
//  - K % 32 == 0: unconditional fragment loads, no selects.
//  - else (K=15): clamped-address loads + value select (no EXEC divergence).
// Fragment layouts per CDNA5 ISA 7.12.2 (wave32).
// ---------------------------------------------------------------------------
template <int K>
__global__ __launch_bounds__(32) void k_wmma_gemm(
    const float* __restrict__ W, const float* __restrict__ bias,
    const float* __restrict__ X, float* __restrict__ Y,
    int M, int N) {
  const int n0   = blockIdx.x * 16;
  const int m0   = blockIdx.y * 16;
  const int b    = blockIdx.z;
  const int lane = (int)threadIdx.x & 31;
  const int grp  = lane >> 4;     // half-wave group
  const int lr   = lane & 15;
  const float* __restrict__ Wr = W + (size_t)(m0 + lr) * K;   // A row for this lane
  const float* __restrict__ Xc = X + (size_t)b * K * N + (n0 + lr);  // B col
  v8f acc = {};
  for (int kk = 0; kk < K; kk += 32) {
    v16h a, bm;
#pragma unroll
    for (int i = 0; i < 16; ++i) {     // A: 16x32 f16, row m=lr
      const int v  = i >> 1, p = i & 1;
      const int kb = (v < 4) ? (2 * v) : (16 + 2 * (v - 4));
      const int k  = kk + kb + p + 8 * grp;
      if (K % 32 == 0) {
        a[i] = (_Float16)Wr[k];
      } else {
        const int ks = (k < K) ? k : 0;          // clamped: load is unconditional
        const float wv = Wr[ks];
        a[i] = (k < K) ? (_Float16)wv : (_Float16)0.0f;  // value select only
      }
    }
#pragma unroll
    for (int i = 0; i < 16; ++i) {     // B: 32x16 f16, col n=lr
      const int k = kk + ((i < 8) ? i : (i + 8)) + 8 * grp;
      if (K % 32 == 0) {
        bm[i] = (_Float16)Xc[(size_t)k * N];
      } else {
        const int ks = (k < K) ? k : 0;
        const float xv = Xc[(size_t)ks * N];
        bm[i] = (k < K) ? (_Float16)xv : (_Float16)0.0f;
      }
    }
    acc = __builtin_amdgcn_wmma_f32_16x16x32_f16(false, a, false, bm,
                                                 (short)0, acc, false, false);
  }
#pragma unroll
  for (int j = 0; j < 8; ++j) {
    const int m = m0 + j + 8 * grp;
    Y[((size_t)b * M + m) * N + (n0 + lr)] = acc[j] + bias[m];
  }
}

// ---------------------------------------------------------------------------
// WMMA conv2d 1xJ over [B,256,16,Win] -> [B,256,16,Wout], weights [256,256,1,J]
// Win/J compile-time so the tap loop unrolls into back-to-back WMMAs.
// ---------------------------------------------------------------------------
template <int Win, int J>
__global__ __launch_bounds__(32) void k_wmma_conv1xj(
    const float* __restrict__ X, const float* __restrict__ W,
    const float* __restrict__ bias, float* __restrict__ Y) {
  const int Wout = Win - J + 1;
  const int n0   = blockIdx.x * 16;   // flattened (r, w)
  const int m0   = blockIdx.y * 16;   // out channel tile
  const int b    = blockIdx.z;
  const int lane = (int)threadIdx.x & 31;
  const int grp  = lane >> 4;
  const int lr   = lane & 15;
  const int nb   = n0 + lr;
  const int rb   = nb / Wout, wb = nb % Wout;
  const float* __restrict__ Wr = W + (size_t)(m0 + lr) * FD * J;
  const float* __restrict__ Xb = X + ((size_t)b * FD * RREG + rb) * Win + wb;
  v8f acc = {};
  for (int c0 = 0; c0 < FD; c0 += 32) {
#pragma unroll
    for (int jj = 0; jj < J; ++jj) {
      v16h a, bm;
#pragma unroll
      for (int i = 0; i < 16; ++i) {
        const int v  = i >> 1, p = i & 1;
        const int kb = (v < 4) ? (2 * v) : (16 + 2 * (v - 4));
        const int c  = c0 + kb + p + 8 * grp;
        a[i] = (_Float16)Wr[(size_t)c * J + jj];
      }
#pragma unroll
      for (int i = 0; i < 16; ++i) {
        const int c = c0 + ((i < 8) ? i : (i + 8)) + 8 * grp;
        bm[i] = (_Float16)Xb[(size_t)c * (RREG * Win) + jj];
      }
      acc = __builtin_amdgcn_wmma_f32_16x16x32_f16(false, a, false, bm,
                                                   (short)0, acc, false, false);
    }
  }
#pragma unroll
  for (int j = 0; j < 8; ++j) {
    const int m = m0 + j + 8 * grp;
    Y[(((size_t)(b * FD + m)) * RREG + rb) * Wout + wb] = acc[j] + bias[m];
  }
}

// ---------------------------------------------------------------------------
// BatchNorm batch-stats over (B, N) of [B,C,N]; one block per channel.
// ---------------------------------------------------------------------------
__global__ __launch_bounds__(256) void k_bn_stats(
    const float* __restrict__ x, float* __restrict__ mu, float* __restrict__ rs,
    int C, int N) {
  const int c = blockIdx.x;
  float s = 0.f, s2 = 0.f;
  for (int i = (int)threadIdx.x; i < Bv * N; i += 256) {
    const int b = i / N, n = i % N;
    const float v = x[((size_t)b * C + c) * N + n];
    s += v; s2 += v * v;
  }
  __shared__ float ls[256], ls2[256];
  ls[threadIdx.x] = s; ls2[threadIdx.x] = s2;
  __syncthreads();
  for (int o = 128; o > 0; o >>= 1) {
    if ((int)threadIdx.x < o) {
      ls[threadIdx.x]  += ls[threadIdx.x + o];
      ls2[threadIdx.x] += ls2[threadIdx.x + o];
    }
    __syncthreads();
  }
  if (threadIdx.x == 0) {
    const float inv = 1.0f / (float)(Bv * N);
    const float m   = ls[0] * inv;
    const float var = ls2[0] * inv - m * m;
    mu[c] = m;
    rs[c] = rsqrtf(var + EPSV);
  }
}

__global__ __launch_bounds__(256) void k_bn_apply(
    float* __restrict__ x, const float* __restrict__ mu, const float* __restrict__ rs,
    const float* __restrict__ g, const float* __restrict__ be,
    int C, int N, int total, int relu) {
  const int i = (int)(blockIdx.x * 256 + threadIdx.x);
  if (i >= total) return;
  const int c = (i / N) % C;
  float v = g[c] * (x[i] - mu[c]) * rs[c] + be[c];
  if (relu) v = fmaxf(v, 0.0f);
  x[i] = v;
}

// max over N: [B,C,N] -> [B,C]; one block per (b,c) row
__global__ __launch_bounds__(256) void k_maxn(const float* __restrict__ x,
                                              float* __restrict__ y, int N) {
  __shared__ float s[256];
  const float* row = x + (size_t)blockIdx.x * N;
  float m = -3.402823466e38f;
  for (int i = (int)threadIdx.x; i < N; i += 256) m = fmaxf(m, row[i]);
  s[threadIdx.x] = m;
  __syncthreads();
  for (int o = 128; o > 0; o >>= 1) {
    if ((int)threadIdx.x < o) s[threadIdx.x] = fmaxf(s[threadIdx.x], s[threadIdx.x + o]);
    __syncthreads();
  }
  if (threadIdx.x == 0) y[blockIdx.x] = s[0];
}

// FC with batch(=2) BN + optional ReLU: out[b,m] = act(bn(in[b,:] . W[m,:] + bias[m]))
// flags: bit0 = BN, bit1 = ReLU
__global__ void k_fc(const float* __restrict__ in, const float* __restrict__ W,
                     const float* __restrict__ bias, const float* __restrict__ g,
                     const float* __restrict__ be, float* __restrict__ out,
                     int M, int K, int flags) {
  const int m = (int)(blockIdx.x * blockDim.x + threadIdx.x);
  if (m >= M) return;
  float y0 = bias[m], y1 = bias[m];
  for (int k = 0; k < K; ++k) {
    const float wv = W[(size_t)m * K + k];
    y0 += in[k] * wv;
    y1 += in[K + k] * wv;
  }
  if (flags & 1) {
    const float mean = 0.5f * (y0 + y1);
    const float d0 = y0 - mean, d1 = y1 - mean;
    const float var = 0.5f * (d0 * d0 + d1 * d1);
    const float r = rsqrtf(var + EPSV);
    y0 = g[m] * d0 * r + be[m];
    y1 = g[m] * d1 * r + be[m];
  }
  if (flags & 2) { y0 = fmaxf(y0, 0.f); y1 = fmaxf(y1, 0.f); }
  out[m]     = y0;
  out[M + m] = y1;
}

// trans = fc3 + eye(15); write to ws and to d_out tail
__global__ void k_make_trans(const float* __restrict__ fc3, float* __restrict__ tws,
                             float* __restrict__ tout) {
  const int i = (int)(blockIdx.x * blockDim.x + threadIdx.x);
  if (i >= Bv * 225) return;
  const int j = i % 225;
  const float v = fc3[i] + (((j % 16) == 0) ? 1.0f : 0.0f);
  tws[i] = v; tout[i] = v;
}

// xt[b,k,n] = sum_c x[b,c,n] * trans[b,c,k]
__global__ __launch_bounds__(256) void k_apply_trans(
    const float* __restrict__ x, const float* __restrict__ tr, float* __restrict__ xt) {
  const int i = (int)(blockIdx.x * 256 + threadIdx.x);
  if (i >= Bv * CIN * NPTS) return;
  const int n = i % NPTS;
  const int k = (i / NPTS) % CIN;
  const int b = i / (NPTS * CIN);
  float s = 0.f;
#pragma unroll
  for (int c = 0; c < CIN; ++c)
    s += x[((size_t)b * CIN + c) * NPTS + n] * tr[b * 225 + c * CIN + k];
  xt[i] = s;
}

// ---------------------------------------------------------------------------
// Per-(b,r) stable descending argsort of 2048 scores (bitonic, in LDS).
// Tie-break: ascending original index (matches jnp.argsort stability on -s).
// ---------------------------------------------------------------------------
__global__ __launch_bounds__(256) void k_argsort(const float* __restrict__ scores,
                                                 int* __restrict__ idxout) {
  __shared__ float ks[NPTS];
  __shared__ int   vi[NPTS];
  const size_t base = (size_t)blockIdx.x * NPTS;
  for (int i = (int)threadIdx.x; i < NPTS; i += 256) { ks[i] = scores[base + i]; vi[i] = i; }
  __syncthreads();
  for (int k = 2; k <= NPTS; k <<= 1) {
    for (int j = k >> 1; j > 0; j >>= 1) {
      for (int i = (int)threadIdx.x; i < NPTS; i += 256) {
        const int ixj = i ^ j;
        if (ixj > i) {
          const bool up = ((i & k) == 0);
          const float a = ks[i], c = ks[ixj];
          const int ai = vi[i], ci = vi[ixj];
          const bool before = (a > c) || (a == c && ai < ci);
          if (up ? !before : before) {
            ks[i] = c; ks[ixj] = a; vi[i] = ci; vi[ixj] = ai;
          }
        }
      }
      __syncthreads();
    }
  }
  for (int i = (int)threadIdx.x; i < NPTS; i += 256) idxout[base + i] = vi[i];
}

// cabins[b,c,r,j] = max over 128-point bin of gathered f; one block per (b,r)
__global__ __launch_bounds__(256) void k_cabins(
    const float* __restrict__ f, const int* __restrict__ idx,
    float* __restrict__ cab_ws, float* __restrict__ cab_out) {
  __shared__ int li[NPTS];
  const int b = (int)blockIdx.x >> 4, r = (int)blockIdx.x & 15, c = (int)threadIdx.x;
  for (int i = (int)threadIdx.x; i < NPTS; i += 256)
    li[i] = idx[((size_t)(b * RREG + r)) * NPTS + i];
  __syncthreads();
  const float* frow = f + ((size_t)(b * FD + c)) * NPTS;
  for (int j = 0; j < NCAB; ++j) {
    float m = -3.402823466e38f;
    for (int p = 0; p < PCAB; ++p) m = fmaxf(m, frow[li[j * PCAB + p]]);
    const size_t o = (((size_t)(b * FD + c)) * RREG + r) * NCAB + j;
    cab_ws[o] = m; cab_out[o] = m;
  }
}

// station[b,o] = bias5[o] + sum_{c,r} trains[b,c,r] * W5[o,c,r]
__global__ __launch_bounds__(256) void k_station(
    const float* __restrict__ trn, const float* __restrict__ W5,
    const float* __restrict__ b5, float* __restrict__ out) {
  const int t = (int)(blockIdx.x * 256 + threadIdx.x);
  if (t >= Bv * FD) return;
  const int b = t / FD, o = t % FD;
  float acc = b5[o];
  for (int c = 0; c < FD; ++c)
#pragma unroll 4
    for (int r = 0; r < RREG; ++r)
      acc += trn[((size_t)(b * FD + c)) * RREG + r] * W5[((size_t)(o * FD + c)) * RREG + r];
  out[t] = acc;
}

// ---------------------------------------------------------------------------
// Emit: gather/broadcast all 1039 feature channels + sp_idx.
// Block = (b, r, channel-group of 32); f rows staged through LDS (async path).
// ---------------------------------------------------------------------------
__global__ __launch_bounds__(256) void k_emit(
    const float* __restrict__ f, const float* __restrict__ xt,
    const int* __restrict__ idx, const float* __restrict__ cab,
    const float* __restrict__ trn, const float* __restrict__ stat,
    float* __restrict__ feat, float* __restrict__ spidx) {
  __shared__ int   li[NPTS];
  __shared__ float row[NPTS];
  const int blk = (int)blockIdx.x;
  const int cg = blk & 7;
  const int r  = (blk >> 3) & 15;
  const int b  = blk >> 7;
  const int tid = (int)threadIdx.x;
  const size_t sect = (size_t)FD * RREG * NPTS;  // 256 channels of feature
  for (int i = tid; i < NPTS; i += 256) li[i] = idx[((size_t)(b * RREG + r)) * NPTS + i];
  __syncthreads();
  for (int cc = 0; cc < 32; ++cc) {
    const int c = cg * 32 + cc;
    stage_row2048(f + ((size_t)(b * FD + c)) * NPTS, row);
    wait_stage_and_barrier();
    const float tv = trn[(b * FD + c) * RREG + r];
    const float sv = stat[b * FD + c];
    const float* cr = cab + (((size_t)(b * FD + c)) * RREG + r) * NCAB;
    for (int i = tid; i < NPTS; i += 256) {
      const int my = li[i];
      const size_t base = (((size_t)(b * 1039 + c)) * RREG + r) * NPTS + i;
      feat[base]            = row[my];       // sp_cube
      feat[base + sect]     = cr[i >> 7];    // sp_windows
      feat[base + 2 * sect] = tv;            // sp_trains
      feat[base + 3 * sect] = sv;            // sp_station
    }
    __syncthreads();  // protect LDS row before next stage
  }
  if (cg == 0) {
    for (int c2 = 0; c2 < CIN; ++c2) {
      stage_row2048(xt + ((size_t)(b * CIN + c2)) * NPTS, row);
      wait_stage_and_barrier();
      for (int i = tid; i < NPTS; i += 256) {
        const int my = li[i];
        feat[(((size_t)(b * 1039 + 1024 + c2)) * RREG + r) * NPTS + i] = row[my];
        spidx[(((size_t)(b * CIN + c2)) * RREG + r) * NPTS + i] = (float)my;
      }
      __syncthreads();
    }
  }
}

// ---------------------------------------------------------------------------
// Host launcher
// ---------------------------------------------------------------------------
enum {
  P_STN_W1, P_STN_B1, P_STN_W2, P_STN_B2, P_STN_W3, P_STN_B3,
  P_STN_FW1, P_STN_FB1, P_STN_FW2, P_STN_FB2, P_STN_FW3, P_STN_FB3,
  P_STN_G1, P_STN_BE1, P_STN_G2, P_STN_BE2, P_STN_G3, P_STN_BE3,
  P_STN_G4, P_STN_BE4, P_STN_G5, P_STN_BE5,
  P_W1, P_B1, P_W2, P_B2, P_W3, P_B3,
  P_G1, P_BE1, P_G2, P_BE2, P_G3, P_BE3,
  P_SORT_W, P_SORT_B,
  P_C1W, P_C1B, P_C2W, P_C2B, P_C3W, P_C3B, P_C4W, P_C4B, P_C5W, P_C5B,
  P_COUNT
};

extern "C" void kernel_launch(void* const* d_in, const int* in_sizes, int n_in,
                              void* d_out, int out_size, void* d_ws, size_t ws_size,
                              hipStream_t stream) {
  (void)out_size; (void)ws_size;
  // Resolve inputs: dict-insertion order (x first) or pytree-alpha order (x last)
  const float* x = nullptr;
  const float* P[P_COUNT];
  if (n_in >= 1 && in_sizes[0] == Bv * CIN * NPTS) {
    x = (const float*)d_in[0];
    for (int i = 0; i < P_COUNT; ++i) P[i] = (const float*)d_in[1 + i];
  } else {
    static const int amap[P_COUNT] = {
      40, 21, 41, 22, 42, 23,           // stn_w1..stn_b3
      32, 29, 33, 30, 34, 31,           // stn_fw1..stn_fb3
      35, 24, 36, 25, 37, 26, 38, 27, 39, 28,  // stn_g1/be1..g5/be5
      43, 0, 44, 1, 45, 2,              // w1,b1,w2,b2,w3,b3
      16, 3, 17, 4, 18, 5,              // g1,be1..g3,be3
      20, 19,                           // sort_w, sort_b
      7, 6, 9, 8, 11, 10, 13, 12, 15, 14  // c2d1..c2d5 (w,b)
    };
    x = (const float*)d_in[n_in - 1];
    for (int i = 0; i < P_COUNT; ++i) P[i] = (const float*)d_in[amap[i]];
  }

  // Workspace carve-up (floats, 256B-aligned chunks)
  float* w = (float*)d_ws;
  size_t off = 0;
  auto alloc = [&](size_t n) { float* p = w + off; off += (n + 63) & ~(size_t)63; return p; };
  float* t64    = alloc((size_t)Bv * 64 * NPTS);
  float* t128   = alloc((size_t)Bv * 128 * NPTS);
  float* t1024  = alloc((size_t)Bv * 1024 * NPTS);
  float* pooled = alloc(Bv * 1024);
  float* fc1    = alloc(Bv * 512);
  float* fc2    = alloc(Bv * 256);
  float* fc3    = alloc(Bv * 225);
  float* twsp   = alloc(Bv * 225);
  float* xt     = alloc((size_t)Bv * CIN * NPTS);
  float* f64    = alloc((size_t)Bv * 64 * NPTS);
  float* f128   = alloc((size_t)Bv * 128 * NPTS);
  float* f256   = alloc((size_t)Bv * FD * NPTS);
  float* scores = alloc((size_t)Bv * RREG * NPTS);
  int*   idxb   = (int*)alloc((size_t)Bv * RREG * NPTS);
  float* cabws  = alloc((size_t)Bv * FD * RREG * NCAB);
  float* tc1    = alloc((size_t)Bv * FD * RREG * 12);
  float* tc2    = alloc((size_t)Bv * FD * RREG * 8);
  float* tc3    = alloc((size_t)Bv * FD * RREG * 4);
  float* trn    = alloc((size_t)Bv * FD * RREG);
  float* stat   = alloc(Bv * FD);
  float* mu     = alloc(1024);
  float* rs     = alloc(1024);

  // d_out sections: feature | cabins | sp_idx | trans
  float* feat_o  = (float*)d_out;
  float* cab_o   = feat_o + (size_t)Bv * 1039 * RREG * NPTS;
  float* spidx_o = cab_o + (size_t)Bv * FD * RREG * NCAB;
  float* trans_o = spidx_o + (size_t)Bv * CIN * RREG * NPTS;

  const dim3 wv(32);
  // ---- STNkd ----
  k_wmma_gemm<15><<<dim3(NPTS / 16, 64 / 16, Bv), wv, 0, stream>>>(P[P_STN_W1], P[P_STN_B1], x, t64, 64, NPTS);
  k_bn_stats<<<64, 256, 0, stream>>>(t64, mu, rs, 64, NPTS);
  k_bn_apply<<<(Bv * 64 * NPTS + 255) / 256, 256, 0, stream>>>(t64, mu, rs, P[P_STN_G1], P[P_STN_BE1], 64, NPTS, Bv * 64 * NPTS, 1);
  k_wmma_gemm<64><<<dim3(NPTS / 16, 128 / 16, Bv), wv, 0, stream>>>(P[P_STN_W2], P[P_STN_B2], t64, t128, 128, NPTS);
  k_bn_stats<<<128, 256, 0, stream>>>(t128, mu, rs, 128, NPTS);
  k_bn_apply<<<(Bv * 128 * NPTS + 255) / 256, 256, 0, stream>>>(t128, mu, rs, P[P_STN_G2], P[P_STN_BE2], 128, NPTS, Bv * 128 * NPTS, 1);
  k_wmma_gemm<128><<<dim3(NPTS / 16, 1024 / 16, Bv), wv, 0, stream>>>(P[P_STN_W3], P[P_STN_B3], t128, t1024, 1024, NPTS);
  k_bn_stats<<<1024, 256, 0, stream>>>(t1024, mu, rs, 1024, NPTS);
  k_bn_apply<<<(Bv * 1024 * NPTS + 255) / 256, 256, 0, stream>>>(t1024, mu, rs, P[P_STN_G3], P[P_STN_BE3], 1024, NPTS, Bv * 1024 * NPTS, 1);
  k_maxn<<<Bv * 1024, 256, 0, stream>>>(t1024, pooled, NPTS);
  k_fc<<<(512 + 63) / 64, 64, 0, stream>>>(pooled, P[P_STN_FW1], P[P_STN_FB1], P[P_STN_G4], P[P_STN_BE4], fc1, 512, 1024, 3);
  k_fc<<<(256 + 63) / 64, 64, 0, stream>>>(fc1, P[P_STN_FW2], P[P_STN_FB2], P[P_STN_G5], P[P_STN_BE5], fc2, 256, 512, 3);
  k_fc<<<(225 + 63) / 64, 64, 0, stream>>>(fc2, P[P_STN_FW3], P[P_STN_FB3], P[P_STN_G5], P[P_STN_BE5], fc3, 225, 256, 0);
  k_make_trans<<<(Bv * 225 + 255) / 256, 256, 0, stream>>>(fc3, twsp, trans_o);
  k_apply_trans<<<(Bv * CIN * NPTS + 255) / 256, 256, 0, stream>>>(x, twsp, xt);

  // ---- feature convs ----
  k_wmma_gemm<15><<<dim3(NPTS / 16, 64 / 16, Bv), wv, 0, stream>>>(P[P_W1], P[P_B1], xt, f64, 64, NPTS);
  k_bn_stats<<<64, 256, 0, stream>>>(f64, mu, rs, 64, NPTS);
  k_bn_apply<<<(Bv * 64 * NPTS + 255) / 256, 256, 0, stream>>>(f64, mu, rs, P[P_G1], P[P_BE1], 64, NPTS, Bv * 64 * NPTS, 1);
  k_wmma_gemm<64><<<dim3(NPTS / 16, 128 / 16, Bv), wv, 0, stream>>>(P[P_W2], P[P_B2], f64, f128, 128, NPTS);
  k_bn_stats<<<128, 256, 0, stream>>>(f128, mu, rs, 128, NPTS);
  k_bn_apply<<<(Bv * 128 * NPTS + 255) / 256, 256, 0, stream>>>(f128, mu, rs, P[P_G2], P[P_BE2], 128, NPTS, Bv * 128 * NPTS, 1);
  k_wmma_gemm<128><<<dim3(NPTS / 16, FD / 16, Bv), wv, 0, stream>>>(P[P_W3], P[P_B3], f128, f256, FD, NPTS);
  k_bn_stats<<<FD, 256, 0, stream>>>(f256, mu, rs, FD, NPTS);
  k_bn_apply<<<(Bv * FD * NPTS + 255) / 256, 256, 0, stream>>>(f256, mu, rs, P[P_G3], P[P_BE3], FD, NPTS, Bv * FD * NPTS, 0);

  // ---- softpool: scores, argsort, cabins ----
  k_wmma_gemm<256><<<dim3(NPTS / 16, RREG / 16, Bv), wv, 0, stream>>>(P[P_SORT_W], P[P_SORT_B], f256, scores, RREG, NPTS);
  k_argsort<<<Bv * RREG, 256, 0, stream>>>(scores, idxb);
  k_cabins<<<Bv * RREG, 256, 0, stream>>>(f256, idxb, cabws, cab_o);

  // ---- conv2d chain ----
  k_wmma_conv1xj<16, 5><<<dim3((RREG * 12) / 16, FD / 16, Bv), wv, 0, stream>>>(cabws, P[P_C1W], P[P_C1B], tc1);
  k_wmma_conv1xj<12, 5><<<dim3((RREG * 8) / 16, FD / 16, Bv), wv, 0, stream>>>(tc1, P[P_C2W], P[P_C2B], tc2);
  k_wmma_conv1xj<8, 5><<<dim3((RREG * 4) / 16, FD / 16, Bv), wv, 0, stream>>>(tc2, P[P_C3W], P[P_C3B], tc3);
  k_wmma_conv1xj<4, 4><<<dim3((RREG * 1) / 16, FD / 16, Bv), wv, 0, stream>>>(tc3, P[P_C4W], P[P_C4B], trn);
  k_station<<<(Bv * FD + 255) / 256, 256, 0, stream>>>(trn, P[P_C5W], P[P_C5B], stat);

  // ---- emit feature + sp_idx (store-bandwidth bound) ----
  k_emit<<<Bv * RREG * 8, 256, 0, stream>>>(f256, xt, idxb, cabws, trn, stat, feat_o, spidx_o);
}
